// MultiHeadAttention_66254165508300
// MI455X (gfx1250) — matrix operand order repaired and checked by reading, not compiled, not run on hardware
//
#include <hip/hip_runtime.h>

// Problem constants
#define H_     2048
#define NH_    16
#define HD_    128
#define BATCH_ 2
#define SEQ_   2048
#define MROWS_ (BATCH_*SEQ_)   // 4096
#define NQKV_  (3*H_)          // 6144

typedef __attribute__((ext_vector_type(16))) __bf16        v16bf;
typedef __attribute__((ext_vector_type(8)))  float         v8f;
typedef __attribute__((ext_vector_type(16))) unsigned short v16us;
typedef __attribute__((ext_vector_type(8)))  unsigned short v8us;
typedef __attribute__((ext_vector_type(4)))  unsigned short v4us;

static __device__ inline unsigned short f2bf(float f) {
  unsigned u = __builtin_bit_cast(unsigned, f);
  u += 0x7FFFu + ((u >> 16) & 1u);          // round-to-nearest-even
  return (unsigned short)(u >> 16);
}

static __device__ inline v8f wmma_bf16(v16bf a, v16bf b, v8f c) {
  return __builtin_amdgcn_wmma_f32_16x16x32_bf16(
      false, a, false, b, (short)0, c, false, false);
}

// A-operand frag: lane holds row m = lane&15; kh = lane>>4 selects
// K halves {kh*8 .. kh*8+7} and {16+kh*8 .. 23+kh*8} (ISA 16-bit A k-map).
static __device__ inline v16bf frag_a(const unsigned short* rowp, int kh) {
  v8us lo = *(const v8us*)(rowp + kh * 8);
  v8us hi = *(const v8us*)(rowp + 16 + kh * 8);
  v16us c;
#pragma unroll
  for (int i = 0; i < 8; ++i) { c[i] = lo[i]; c[i + 8] = hi[i]; }
  return __builtin_bit_cast(v16bf, c);
}
// B-operand frag: lane holds col n = lane&15; K halves {kh*8 .. kh*8+7} and {16+kh*8 .. 23+kh*8}.
static __device__ inline v16bf frag_b(const unsigned short* rowp, int kh) {
  v8us lo = *(const v8us*)(rowp + kh * 8);
  v8us hi = *(const v8us*)(rowp + 16 + kh * 8);
  v16us c;
#pragma unroll
  for (int i = 0; i < 8; ++i) { c[i] = lo[i]; c[i + 8] = hi[i]; }
  return __builtin_bit_cast(v16bf, c);
}

// ---- CDNA5 async memory->LDS DMA (GVS addressing: SGPR base + VGPR i32 offset)
// VDST VGPR carries the LDS byte address (low 32 bits of generic LDS pointer).
static __device__ inline void async_ld_b128(unsigned lds_byte_addr,
                                            const void* sbase, int voff_bytes) {
  asm volatile("global_load_async_to_lds_b128 %0, %1, %2"
               :: "v"(lds_byte_addr), "v"(voff_bytes), "s"(sbase)
               : "memory");
}
static __device__ inline void wait_async_le(int n) {
  if (n == 0)
    asm volatile("s_wait_asynccnt 0x0" ::: "memory");
  else
    asm volatile("s_wait_asynccnt 0x4" ::: "memory");
}

// ---------------- f32 -> bf16 conversion ----------------
__global__ __launch_bounds__(256)
void cvt_bf16(const float* __restrict__ in, unsigned short* __restrict__ out, int n) {
  int i = (blockIdx.x * 256 + threadIdx.x) * 4;
  if (i < n) {
    float4 v = *(const float4*)(in + i);
    v4us r;
    r[0] = f2bf(v.x); r[1] = f2bf(v.y); r[2] = f2bf(v.z); r[3] = f2bf(v.w);
    *(v4us*)(out + i) = r;
  }
}

// ---------------- GEMM: C(MxN) = A(MxK) * W(NxK)^T + bias ----------------
// Double-buffered: async DMA fills the next 128x32 A/W slab in LDS while the
// current slab feeds 8 WMMAs per wave per K-step.
// MODE 0: QKV projection, epilogue scatters bf16 into Q(B,N,S,D), K(B,N,S,D), Vt(B,N,D,S)
// MODE 1: output projection, epilogue writes f32 to outF
template <int MODE>
__global__ __launch_bounds__(256)
void gemm_xwT(const unsigned short* __restrict__ A,
              const unsigned short* __restrict__ W,
              const float* __restrict__ bias,
              float* __restrict__ outF,
              unsigned short* __restrict__ qout,
              unsigned short* __restrict__ kout,
              unsigned short* __restrict__ vtout,
              int Ndim, int Kdim) {
  __shared__ unsigned short lA[2][128 * 40];
  __shared__ unsigned short lW[2][128 * 40];
  const int tid  = threadIdx.x;
  const int lane = tid & 31;
  const int w    = tid >> 5;   // 8 waves
  const int wm   = w & 1;      // 2 wave-rows of 64
  const int wn   = w >> 1;     // 4 wave-cols of 32
  const int row0 = blockIdx.y * 128;
  const int col0 = blockIdx.x * 128;
  const int mrow = lane & 15;
  const int kh   = lane >> 4;

  v8f acc[4][2];
#pragma unroll
  for (int mt = 0; mt < 4; ++mt)
#pragma unroll
    for (int nt = 0; nt < 2; ++nt)
#pragma unroll
      for (int i = 0; i < 8; ++i) acc[mt][nt][i] = 0.f;

  // each thread DMAs one 16-elem half-row of A and of W per K-step (2x b128 each)
  const int lrow = tid >> 1;          // 0..127
  const int lcol = (tid & 1) * 16;    // 0 or 16
  const unsigned ldsA0 = (unsigned)(unsigned long long)(const void*)&lA[0][lrow * 40 + lcol];
  const unsigned ldsW0 = (unsigned)(unsigned long long)(const void*)&lW[0][lrow * 40 + lcol];
  const unsigned bufB  = 128 * 40 * 2;          // bytes per buffer
  const int aoff0 = ((row0 + lrow) * Kdim + lcol) * 2;  // byte offsets
  const int woff0 = ((col0 + lrow) * Kdim + lcol) * 2;

#define ISSUE_STEP(step, buf)                                              \
  do {                                                                     \
    unsigned la = ldsA0 + (unsigned)(buf) * bufB;                          \
    unsigned lw = ldsW0 + (unsigned)(buf) * bufB;                          \
    int ka = aoff0 + (step) * 64;                                          \
    int kw = woff0 + (step) * 64;                                          \
    async_ld_b128(la,      A, ka);                                         \
    async_ld_b128(la + 16, A, ka + 16);                                    \
    async_ld_b128(lw,      W, kw);                                         \
    async_ld_b128(lw + 16, W, kw + 16);                                    \
  } while (0)

  const int steps = Kdim / 32;
  ISSUE_STEP(0, 0);

  for (int i = 0; i < steps; ++i) {
    const int cur = i & 1;
    if (i + 1 < steps) {
      ISSUE_STEP(i + 1, cur ^ 1);
      wait_async_le(4);     // older (current-slab) DMAs complete, next slab in flight
    } else {
      wait_async_le(0);
    }
    __syncthreads();        // slab visible to all waves

    v16bf bfr[2];
#pragma unroll
    for (int nt = 0; nt < 2; ++nt)
      bfr[nt] = frag_b(&lW[cur][(wn * 32 + nt * 16 + mrow) * 40], kh);
#pragma unroll
    for (int mt = 0; mt < 4; ++mt) {
      v16bf afr = frag_a(&lA[cur][(wm * 64 + mt * 16 + mrow) * 40], kh);
#pragma unroll
      for (int nt = 0; nt < 2; ++nt)
        acc[mt][nt] = wmma_bf16(afr, bfr[nt], acc[mt][nt]);
    }
    __syncthreads();        // all reads of `cur` done before it is re-filled
  }
#undef ISSUE_STEP

  // Epilogue: C layout is VGPR r, lanes0-15 -> M=r, lanes16-31 -> M=r+8; N = lane&15
  const int rofs = kh * 8;
#pragma unroll
  for (int mt = 0; mt < 4; ++mt) {
#pragma unroll
    for (int nt = 0; nt < 2; ++nt) {
#pragma unroll
      for (int r = 0; r < 8; ++r) {
        int grow = row0 + wm * 64 + mt * 16 + r + rofs;
        int gcol = col0 + wn * 32 + nt * 16 + mrow;
        float v = acc[mt][nt][r] + bias[gcol];
        if (MODE == 0) {
          int which = gcol >> 11;       // 0=q 1=k 2=v
          int rem   = gcol & 2047;
          int hh = rem >> 7, dd = rem & 127;
          int bb = grow >> 11, ss = grow & 2047;
          size_t bh = (size_t)bb * NH_ + hh;
          unsigned short bv = f2bf(v);
          if (which == 0)      qout[(bh * SEQ_ + ss) * HD_ + dd] = bv;
          else if (which == 1) kout[(bh * SEQ_ + ss) * HD_ + dd] = bv;
          else                 vtout[(bh * HD_ + dd) * SEQ_ + ss] = bv;
        } else {
          outF[(size_t)grow * Ndim + gcol] = v;
        }
      }
    }
  }
}

// ---------------- Flash attention (causal) ----------------
// grid: (SEQ/64, NH, B), block: 128 (4 waves). Each wave owns 16 q rows.
__global__ __launch_bounds__(128)
void flash_attn(const unsigned short* __restrict__ Qb,
                const unsigned short* __restrict__ Kb,
                const unsigned short* __restrict__ Vt,
                unsigned short* __restrict__ attO) {
  __shared__ unsigned short lp[4][16 * 40];   // per-wave P tile 16x32 (pad 8)
  const int lane = threadIdx.x & 31;
  const int w    = threadIdx.x >> 5;
  const int mrow = lane & 15;
  const int kh   = lane >> 4;
  const int bh   = blockIdx.z * NH_ + blockIdx.y;
  const int q0   = blockIdx.x * 64 + w * 16;

  // Q fragments resident in VGPRs (16 rows x 128 d)
  const unsigned short* Qrow = Qb + ((size_t)bh * SEQ_ + q0 + mrow) * HD_;
  v16bf qf[4];
#pragma unroll
  for (int kk = 0; kk < 4; ++kk) qf[kk] = frag_a(Qrow + kk * 32, kh);

  v8f o[8];
  float mrun[8], lrun[8];
#pragma unroll
  for (int nt = 0; nt < 8; ++nt)
#pragma unroll
    for (int i = 0; i < 8; ++i) o[nt][i] = 0.f;
#pragma unroll
  for (int r = 0; r < 8; ++r) { mrun[r] = -1e30f; lrun[r] = 0.f; }

  unsigned short* P = lp[w];
  const float scale = 0.08838834764831845f;   // 1/sqrt(128)
  const int qmax = q0 + 15;

  for (int kb = 0; kb <= qmax; kb += 32) {
    // prefetch next key block's K rows and V columns (global_prefetch_b8)
    if (kb + 32 <= qmax) {
      __builtin_prefetch(Kb + ((size_t)bh * SEQ_ + kb + 32 + mrow) * HD_, 0, 0);
      __builtin_prefetch(Vt + ((size_t)bh * HD_ + mrow) * SEQ_ + kb + 32, 0, 0);
      __builtin_prefetch(Vt + ((size_t)bh * HD_ + 64 + mrow) * SEQ_ + kb + 32, 0, 0);
    }
    v8f sfr[2];
#pragma unroll
    for (int j = 0; j < 2; ++j) {
      v8f s;
#pragma unroll
      for (int i = 0; i < 8; ++i) s[i] = 0.f;
      const unsigned short* Krow =
          Kb + ((size_t)bh * SEQ_ + kb + j * 16 + mrow) * HD_;
#pragma unroll
      for (int kk = 0; kk < 4; ++kk)
        s = wmma_bf16(qf[kk], frag_b(Krow + kk * 32, kh), s);
      int kcol = kb + j * 16 + mrow;
#pragma unroll
      for (int r = 0; r < 8; ++r) {
        int qrow = q0 + r + kh * 8;
        s[r] = (kcol > qrow) ? -1e30f : s[r] * scale;
      }
      sfr[j] = s;
    }
    // online softmax: reductions across the 16-lane half (offsets 1..8 keep halves intact)
    float rmax[8], rsum[8], alpha[8];
#pragma unroll
    for (int r = 0; r < 8; ++r) rmax[r] = fmaxf(sfr[0][r], sfr[1][r]);
#pragma unroll
    for (int off = 1; off < 16; off <<= 1)
#pragma unroll
      for (int r = 0; r < 8; ++r)
        rmax[r] = fmaxf(rmax[r], __shfl_xor(rmax[r], off, 32));
#pragma unroll
    for (int r = 0; r < 8; ++r) {
      float mn = fmaxf(mrun[r], rmax[r]);
      alpha[r] = __expf(mrun[r] - mn);
      mrun[r] = mn;
      rsum[r] = 0.f;
    }
#pragma unroll
    for (int j = 0; j < 2; ++j)
#pragma unroll
      for (int r = 0; r < 8; ++r) {
        float p = __expf(sfr[j][r] - mrun[r]);
        sfr[j][r] = p;
        rsum[r] += p;
      }
#pragma unroll
    for (int off = 1; off < 16; off <<= 1)
#pragma unroll
      for (int r = 0; r < 8; ++r) rsum[r] += __shfl_xor(rsum[r], off, 32);
#pragma unroll
    for (int r = 0; r < 8; ++r) lrun[r] = lrun[r] * alpha[r] + rsum[r];
#pragma unroll
    for (int nt = 0; nt < 8; ++nt)
#pragma unroll
      for (int i = 0; i < 8; ++i) o[nt][i] *= alpha[i];

    // stage P (C-layout -> LDS row-major 16x32), wave-private, LDS is in-order per wave
#pragma unroll
    for (int j = 0; j < 2; ++j)
#pragma unroll
      for (int r = 0; r < 8; ++r)
        P[(r + kh * 8) * 40 + j * 16 + mrow] = f2bf(sfr[j][r]);

    v16bf pa = frag_a(P + mrow * 40, kh);      // A-operand reload of P
#pragma unroll
    for (int nt = 0; nt < 8; ++nt) {
      const unsigned short* Vrow =
          Vt + ((size_t)bh * HD_ + nt * 16 + mrow) * SEQ_ + kb;  // transposed V: contiguous keys
      o[nt] = wmma_bf16(pa, frag_b(Vrow, kh), o[nt]);
    }
  }

  // write att output as bf16 in (B,S,H)
#pragma unroll
  for (int nt = 0; nt < 8; ++nt)
#pragma unroll
    for (int r = 0; r < 8; ++r) {
      int qrow = q0 + r + kh * 8;
      attO[((size_t)blockIdx.z * SEQ_ + qrow) * H_ + blockIdx.y * HD_ + nt * 16 + mrow] =
          f2bf(o[nt][r] / lrun[r]);
    }
}

// ---------------- launch ----------------
extern "C" void kernel_launch(void* const* d_in, const int* in_sizes, int n_in,
                              void* d_out, int out_size, void* d_ws, size_t ws_size,
                              hipStream_t stream) {
  (void)in_sizes; (void)n_in; (void)out_size; (void)ws_size;
  const float* x    = (const float*)d_in[0];
  const float* Wqkv = (const float*)d_in[1];
  const float* bqkv = (const float*)d_in[2];
  const float* Wout = (const float*)d_in[3];
  const float* bout = (const float*)d_in[4];
  float* out = (float*)d_out;

  unsigned short* p   = (unsigned short*)d_ws;
  unsigned short* xb  = p;  p += (size_t)MROWS_ * H_;        // x bf16
  unsigned short* wqb = p;  p += (size_t)NQKV_ * H_;         // W_qkv bf16
  unsigned short* wob = p;  p += (size_t)H_ * H_;            // W_out bf16
  unsigned short* qb  = p;  p += (size_t)BATCH_ * NH_ * SEQ_ * HD_;  // Q (B,N,S,D)
  unsigned short* kbf = p;  p += (size_t)BATCH_ * NH_ * SEQ_ * HD_;  // K (B,N,S,D)
  unsigned short* vt  = p;  p += (size_t)BATCH_ * NH_ * SEQ_ * HD_;  // V (B,N,D,S)
  unsigned short* att = p;  p += (size_t)MROWS_ * H_;        // attention out bf16

  const int nx = MROWS_ * H_, nwq = NQKV_ * H_, nwo = H_ * H_;
  cvt_bf16<<<nx  / 1024, 256, 0, stream>>>(x,    xb,  nx);
  cvt_bf16<<<nwq / 1024, 256, 0, stream>>>(Wqkv, wqb, nwq);
  cvt_bf16<<<nwo / 1024, 256, 0, stream>>>(Wout, wob, nwo);

  gemm_xwT<0><<<dim3(NQKV_ / 128, MROWS_ / 128), 256, 0, stream>>>(
      xb, wqb, bqkv, nullptr, qb, kbf, vt, NQKV_, H_);

  flash_attn<<<dim3(SEQ_ / 64, NH_, BATCH_), 128, 0, stream>>>(qb, kbf, vt, att);

  gemm_xwT<1><<<dim3(H_ / 128, MROWS_ / 128), 256, 0, stream>>>(
      att, wob, bout, out, nullptr, nullptr, nullptr, H_, H_);
}
